// ParamMakeFeature_51556787421342
// MI455X (gfx1250) — matrix operand order repaired and checked by reading, not compiled, not run on hardware
//
#include <hip/hip_runtime.h>
#include <hip/hip_bf16.h>

typedef __attribute__((ext_vector_type(16))) _Float16 v16h;
typedef __attribute__((ext_vector_type(8)))  _Float16 v8h;
typedef __attribute__((ext_vector_type(8)))  float    v8f;
typedef __attribute__((ext_vector_type(4)))  float    v4f;

#define NB 32
#define NPTS 65536

// ---------------------------------------------------------------------------
// Prep kernel: one wave per batch. Builds fused per-batch operands as compact
// 16x16 f16 row-major matrices (row = output feature, col = input feature):
//   A1 = W1 * R^T   (15x3  -> 16x16, cols 3..15 zero)
//   A2 = W2         (15x15 -> 16x16) [batch-independent]
//   A3 = D_b * W3   (15x15 -> 16x16)
//   bias3' = D_b * b3; bias1/bias2 padded to 16 floats
// D_b = blockdiag( R, Q2 (R kron R) Q2^T, Q3 (R kron R kron R) Q3^T ).
//
// The main kernel loads row f of these compact matrices into A-operand
// elements 0..7: features 0..7 in lane f, 8..15 in lane f+16 (elems 8..15 zero). Per the 16-bit 16x32 A
// layout, that maps feature-k to WMMA K = k for all 16 features (K=16..31 zero),
// i.e. exactly the columns that the C->B pass-through layout feeds — so no
// cross-lane movement is needed between layers.
// ---------------------------------------------------------------------------
__global__ __launch_bounds__(32) void rotmlp_prep(
    const float* __restrict__ Rg, const float* __restrict__ W1,
    const float* __restrict__ b1, const float* __restrict__ W2,
    const float* __restrict__ b2, const float* __restrict__ W3,
    const float* __restrict__ b3, const float* __restrict__ Q2,
    const float* __restrict__ Q3,
    _Float16* __restrict__ a1, _Float16* __restrict__ a2,
    _Float16* __restrict__ a3, float* __restrict__ bias1,
    float* __restrict__ bias2, float* __restrict__ bias3)
{
    const int b    = blockIdx.x;
    const int lane = threadIdx.x;   // 0..31

    __shared__ float D[16][16];
    __shared__ float R[9];

    for (int i = lane; i < 256; i += 32) (&D[0][0])[i] = 0.0f;
    if (lane < 9) R[lane] = Rg[b * 9 + lane];
    __syncthreads();

    // l=1 block: D1 = R  (Q1 = I)
    if (lane < 9) D[lane / 3][lane % 3] = R[lane];

    // l=2 block: D2[p][q] = sum_{u,v} Q2[p,u] * (R kron R)[u,v] * Q2[q,v]
    if (lane < 25) {
        const int p = lane / 5, q = lane % 5;
        float acc = 0.0f;
        for (int u = 0; u < 9; ++u) {
            const float qu = Q2[p * 9 + u];
            const int ui = u / 3, uk = u % 3;
            for (int v = 0; v < 9; ++v) {
                const float k2 = R[ui * 3 + (v / 3)] * R[uk * 3 + (v % 3)];
                acc += qu * k2 * Q2[q * 9 + v];
            }
        }
        D[3 + p][3 + q] = acc;
    }

    // l=3 block: D3[p][q] = sum_{u,v} Q3[p,u] * (R kron^2 R)[u,v] * Q3[q,v]
    for (int idx = lane; idx < 49; idx += 32) {
        const int p = idx / 7, q = idx % 7;
        float acc = 0.0f;
        for (int u = 0; u < 27; ++u) {
            const float qu = Q3[p * 27 + u];
            const int ui = u / 9, uk = (u / 3) % 3, um = u % 3;
            for (int v = 0; v < 27; ++v) {
                const float k3 = R[ui * 3 + (v / 9)] *
                                 R[uk * 3 + ((v / 3) % 3)] *
                                 R[um * 3 + (v % 3)];
                acc += qu * k3 * Q3[q * 27 + v];
            }
        }
        D[8 + p][8 + q] = acc;
    }
    __syncthreads();

    if (lane < 16) {
        // A1 = W1 * R^T : A1[f][c] = sum_i W1[f,i] * R[c,i]
        _Float16* d1 = a1 + ((size_t)b * 16 + lane) * 16;
        float row[3] = {0.f, 0.f, 0.f};
        if (lane < 15)
            for (int c = 0; c < 3; ++c) {
                float acc = 0.f;
                for (int i = 0; i < 3; ++i) acc += W1[lane * 3 + i] * R[c * 3 + i];
                row[c] = acc;
            }
        for (int k = 0; k < 16; ++k) d1[k] = (_Float16)((k < 3) ? row[k] : 0.0f);

        // A3 = D * W3
        _Float16* d3 = a3 + ((size_t)b * 16 + lane) * 16;
        for (int k = 0; k < 16; ++k) {
            float acc = 0.f;
            if (lane < 15 && k < 15)
                for (int q = 0; q < 15; ++q) acc += D[lane][q] * W3[q * 15 + k];
            d3[k] = (_Float16)acc;
        }
        // bias3' = D * b3
        float acc = 0.f;
        if (lane < 15)
            for (int q = 0; q < 15; ++q) acc += D[lane][q] * b3[q];
        bias3[b * 16 + lane] = acc;

        if (b == 0) {
            _Float16* d2 = a2 + (size_t)lane * 16;
            for (int k = 0; k < 16; ++k)
                d2[k] = (_Float16)((lane < 15 && k < 15) ? W2[lane * 15 + k] : 0.0f);
            bias1[lane] = (lane < 15) ? b1[lane] : 0.0f;
            bias2[lane] = (lane < 15) ? b2[lane] : 0.0f;
        }
    }
}

// ---------------------------------------------------------------------------
// Main kernel helpers
// ---------------------------------------------------------------------------
__device__ __forceinline__ float bperm_f(float v, int srcLane) {
    return __builtin_bit_cast(float,
        __builtin_amdgcn_ds_bpermute((srcLane & 31) << 2,
                                     __builtin_bit_cast(int, v)));
}

// C -> next-layer B pass-through: every lane just converts its own 8
// accumulator values. With this weight layout, lane n feeds
// K=0..7 (features 0..7 of column n) and lane n+16 feeds K=8..15
// (features 8..15 of column n). K=16..31 columns of A are zero,
// so B elems 8..15 are don't-care (kept zero).
__device__ __forceinline__ v16h pack_b(const float h[8]) {
    v16h bv = {};
#pragma unroll
    for (int r = 0; r < 8; ++r) bv[r] = (_Float16)h[r];
    return bv;
}

// ---------------------------------------------------------------------------
// Main kernel: 8 waves/block, each wave processes 32 points as two 16-pt WMMA
// tiles, interleaved so the second tile's VALU work hides the first tile's
// WMMA->VALU hazard window. Weights stay resident in VGPRs; x streamed once,
// out written once (memory-bound: ~151 MB -> ~6.5 us at 23.3 TB/s).
// ---------------------------------------------------------------------------
__global__ __launch_bounds__(256) void rotmlp_main(
    const float* __restrict__ x,
    const _Float16* __restrict__ a1, const _Float16* __restrict__ a2,
    const _Float16* __restrict__ a3,
    const float* __restrict__ bias1, const float* __restrict__ bias2,
    const float* __restrict__ bias3, float* __restrict__ out)
{
    const int lane = threadIdx.x & 31;
    const int wave = threadIdx.x >> 5;
    const int b    = blockIdx.y;
    const int base = (blockIdx.x * 8 + wave) * 32;

    const int  row = lane & 15;
    const bool act = lane < 16;
    const int  koff = act ? 0 : 8;

    // Stationary weight A-operands: lane l holds features koff..koff+7 of row
    // l%16 in elements 0..7 (one 16B load); elements 8..15 are zero.
    v16h A1 = {}, A2 = {}, A3 = {};
    {
        const v8h t1 = *(const v8h*)(a1 + ((size_t)b * 16 + row) * 16 + koff);
        const v8h t2 = *(const v8h*)(a2 + (size_t)row * 16 + koff);
        const v8h t3 = *(const v8h*)(a3 + ((size_t)b * 16 + row) * 16 + koff);
#pragma unroll
        for (int r = 0; r < 8; ++r) { A1[r] = t1[r]; A2[r] = t2[r]; A3[r] = t3[r]; }
    }

    // Biases for this lane's C rows (rows koff..koff+7)
    const v4f b1lo = *(const v4f*)(bias1 + koff);
    const v4f b1hi = *(const v4f*)(bias1 + koff + 4);
    const v4f b2lo = *(const v4f*)(bias2 + koff);
    const v4f b2hi = *(const v4f*)(bias2 + koff + 4);
    const v4f b3lo = *(const v4f*)(bias3 + b * 16 + koff);
    const v4f b3hi = *(const v4f*)(bias3 + b * 16 + koff + 4);
    float bb1[8], bb2[8], bb3[8];
#pragma unroll
    for (int r = 0; r < 4; ++r) {
        bb1[r] = b1lo[r]; bb1[4 + r] = b1hi[r];
        bb2[r] = b2lo[r]; bb2[4 + r] = b2hi[r];
        bb3[r] = b3lo[r]; bb3[4 + r] = b3hi[r];
    }

    // Stream 32 points of x (3 channels, fully coalesced 128B per channel)
    const float* xb = x + (size_t)b * 3 * NPTS + base;
    const float x0 = xb[lane];
    const float x1 = xb[NPTS + lane];
    const float x2 = xb[2 * NPTS + lane];

    // Tile 1 x values for lanes 0..15 come from lanes 16..31 (3 ds_bpermute;
    // upper-lane results are don't-care: they hit zero A columns).
    const float y0 = bperm_f(x0, lane + 16);
    const float y1 = bperm_f(x1, lane + 16);
    const float y2 = bperm_f(x2, lane + 16);

    // ---- Layer 1 (K=0..2 live; A1 K>=3 columns are zero, so no masking) ----
    v16h B1a = {}, B1b = {};
    B1a[0] = (_Float16)x0; B1a[1] = (_Float16)x1; B1a[2] = (_Float16)x2;
    B1b[0] = (_Float16)y0; B1b[1] = (_Float16)y1; B1b[2] = (_Float16)y2;

    v8f c0 = {}, c1 = {};
    c0 = __builtin_amdgcn_wmma_f32_16x16x32_f16(false, A1, false, B1a,
                                                (short)0, c0, false, false);
    c1 = __builtin_amdgcn_wmma_f32_16x16x32_f16(false, A1, false, B1b,
                                                (short)0, c1, false, false);
    float h0[8], h1[8];
#pragma unroll
    for (int r = 0; r < 8; ++r) h0[r] = fmaxf(c0[r] + bb1[r], 0.0f);
#pragma unroll
    for (int r = 0; r < 8; ++r) h1[r] = fmaxf(c1[r] + bb1[r], 0.0f);

    // ---- Layer 2 ----
    const v16h B2a = pack_b(h0);
    const v16h B2b = pack_b(h1);
    v8f d0 = {}, d1 = {};
    d0 = __builtin_amdgcn_wmma_f32_16x16x32_f16(false, A2, false, B2a,
                                                (short)0, d0, false, false);
    d1 = __builtin_amdgcn_wmma_f32_16x16x32_f16(false, A2, false, B2b,
                                                (short)0, d1, false, false);
#pragma unroll
    for (int r = 0; r < 8; ++r) h0[r] = fmaxf(d0[r] + bb2[r], 0.0f);
#pragma unroll
    for (int r = 0; r < 8; ++r) h1[r] = fmaxf(d1[r] + bb2[r], 0.0f);

    // ---- Layer 3 (rotation D_b folded into A3 / bias3) ----
    const v16h B3a = pack_b(h0);
    const v16h B3b = pack_b(h1);
    v8f e0 = {}, e1 = {};
    e0 = __builtin_amdgcn_wmma_f32_16x16x32_f16(false, A3, false, B3a,
                                                (short)0, e0, false, false);
    e1 = __builtin_amdgcn_wmma_f32_16x16x32_f16(false, A3, false, B3b,
                                                (short)0, e1, false, false);

    // Store: row m = koff + r (skip padded feature row 15); each feature
    // row's 16 lanes write 64B contiguous.
    const int n0 = base + row;
    const int n1 = base + 16 + row;
#pragma unroll
    for (int r = 0; r < 8; ++r) {
        const int m = koff + r;
        if (m < 15) {
            out[((size_t)b * 15 + m) * NPTS + n0] = e0[r] + bb3[r];
            out[((size_t)b * 15 + m) * NPTS + n1] = e1[r] + bb3[r];
        }
    }
}

extern "C" void kernel_launch(void* const* d_in, const int* in_sizes, int n_in,
                              void* d_out, int out_size, void* d_ws, size_t ws_size,
                              hipStream_t stream) {
    const float* x  = (const float*)d_in[0];
    const float* R  = (const float*)d_in[1];
    const float* W1 = (const float*)d_in[2];
    const float* b1 = (const float*)d_in[3];
    const float* W2 = (const float*)d_in[4];
    const float* b2 = (const float*)d_in[5];
    const float* W3 = (const float*)d_in[6];
    const float* b3 = (const float*)d_in[7];
    // d_in[8] = Q1 (identity, unused)
    const float* Q2 = (const float*)d_in[9];
    const float* Q3 = (const float*)d_in[10];
    float* out = (float*)d_out;

    char* ws = (char*)d_ws;
    _Float16* a1    = (_Float16*)(ws);           // 32*16*16*2 = 16384 B
    _Float16* a3    = (_Float16*)(ws + 16384);   // 16384 B
    _Float16* a2    = (_Float16*)(ws + 32768);   // 512 B
    float*    bias1 = (float*)(ws + 33280);      // 64 B
    float*    bias2 = (float*)(ws + 33344);      // 64 B
    float*    bias3 = (float*)(ws + 33408);      // 2048 B

    rotmlp_prep<<<dim3(NB), dim3(32), 0, stream>>>(
        R, W1, b1, W2, b2, W3, b3, Q2, Q3, a1, a2, a3, bias1, bias2, bias3);

    rotmlp_main<<<dim3(NPTS / 256, NB), dim3(256), 0, stream>>>(
        x, a1, a2, a3, bias1, bias2, bias3, out);
}
